// ESN_1262720385325
// MI455X (gfx1250) — compile-verified, matrix operand order checked by reading, not run on hardware
//
#include <hip/hip_runtime.h>
#include <math.h>

// ESN on MI455X (gfx1250, wave32).  B=32, T=512, D=128, N=1024.
//   - bf16x3 split-precision WMMA (v_wmma_f32_16x16x32_bf16) everywhere.
//   - B matrices (w_res, w_input) pre-split into bf16 hi/lo in WMMA fragment
//     layout (32 contiguous bytes per lane -> b128 loads).
//   - Recurrent scan: persistent kernel, 128 co-resident blocks, grid barrier
//     per step. The state is kept as PRE-PACKED bf16 hi/lo A-fragments
//     (double buffered): the epilogue that computes tanh(...) also emits the
//     packed bytes, so the scan's inner loop is pure loads + WMMA.
//   - d_out doubles as the xin buffer: esn_xin writes xin into out[:,t,:],
//     the scan adds state@w_res, applies tanh+noise, overwrites in place.

typedef __attribute__((ext_vector_type(16))) __bf16 v16bf;
typedef __attribute__((ext_vector_type(8)))  float  v8f;
typedef __attribute__((ext_vector_type(8)))  int    v8i;
typedef __attribute__((ext_vector_type(2)))  float  v2f;

#define NOISE_F 0.001f

static __device__ __forceinline__ unsigned pack_bf16_pair(float a, float b) {
  unsigned short ua = __builtin_bit_cast(unsigned short, (__bf16)a);
  unsigned short ub = __builtin_bit_cast(unsigned short, (__bf16)b);
  return ((unsigned)ub << 16) | (unsigned)ua;
}

// Fragment k-pattern shared by A and B layouts:
//   k(v, half) = (v&3)*2 + ((v>>2)<<4) + half*8   (+0 / +1 inside the dword)
// A frag: lane = m | (half<<4);  B frag: lane = n | (half<<4).
// dword index = ((tile*kblks + kb)*32 + lane)*8 + v.

// ---------------------------------------------------------------------------
// Pack f32 row-major [K x N] into bf16 hi/lo B fragments.
// Also zero-inits the global barrier (block 0) when bar != nullptr.
// ---------------------------------------------------------------------------
__global__ void esn_pack(const float* __restrict__ w, int K, int N,
                         unsigned* __restrict__ hi, unsigned* __restrict__ lo,
                         int* bar) {
  if (bar && blockIdx.x == 0 && threadIdx.x < 2) bar[threadIdx.x] = 0;
  const int kblks = K >> 5;
  const int total = (N >> 4) * kblks * 32 * 8;
  int idx = blockIdx.x * blockDim.x + threadIdx.x;
  if (idx >= total) return;
  const int v    = idx & 7;
  const int lane = (idx >> 3) & 31;
  const int kb   = (idx >> 8) % kblks;
  const int nt   = (idx >> 8) / kblks;
  const int n = nt * 16 + (lane & 15);
  const int k = kb * 32 + (v & 3) * 2 + ((v >> 2) << 4) + ((lane >> 4) << 3);
  const float w0 = w[(size_t)k * N + n];
  const float w1 = w[(size_t)(k + 1) * N + n];
  const float h0 = (float)(__bf16)w0;
  const float h1 = (float)(__bf16)w1;
  hi[idx] = pack_bf16_pair(w0, w1);
  lo[idx] = pack_bf16_pair(w0 - h0, w1 - h1);
}

// ---------------------------------------------------------------------------
// Pack states0 [32 x 1024] f32 into bf16 hi/lo A fragments (phase-0 buffer).
// Total dwords: 2 mtiles * 32 kblks * 32 lanes * 8 = 16384.
// ---------------------------------------------------------------------------
__global__ void esn_pack_s0(const float* __restrict__ s0,
                            unsigned* __restrict__ ah, unsigned* __restrict__ al) {
  int idx = blockIdx.x * blockDim.x + threadIdx.x;
  if (idx >= 2 * 32 * 32 * 8) return;
  const int v    = idx & 7;
  const int lane = (idx >> 3) & 31;
  const int kb   = (idx >> 8) & 31;
  const int mt   = idx >> 13;
  const int m = mt * 16 + (lane & 15);
  const int k = kb * 32 + (v & 3) * 2 + ((v >> 2) << 4) + ((lane >> 4) << 3);
  const float f0 = s0[(size_t)m * 1024 + k];
  const float f1 = s0[(size_t)m * 1024 + k + 1];
  const float h0 = (float)(__bf16)f0;
  const float h1 = (float)(__bf16)f1;
  ah[idx] = pack_bf16_pair(f0, f1);
  al[idx] = pack_bf16_pair(f0 - h0, f1 - h1);
}

// ---------------------------------------------------------------------------
// out[:,0,:] = step0
// ---------------------------------------------------------------------------
__global__ void esn_t0(const float* __restrict__ step0, float* __restrict__ out) {
  int idx = blockIdx.x * blockDim.x + threadIdx.x;
  if (idx < 32 * 1024) {
    int b = idx >> 10, n = idx & 1023;
    out[((size_t)b * 512) * 1024 + n] = step0[idx];
  }
}

// ---------------------------------------------------------------------------
// xin: out[b,t,n] = scale * sum_d x[b,t,d]*w_input[d,n], t = 1..511.
// One block per m-tile (1022 blocks). Each wave converts its 16 x-rows ONCE
// into register-resident A fragments (4 kblks x hi/lo), then sweeps 16
// n-tiles: 16 * 4 * 3 = 192 WMMAs per wave against in-register A.
// ---------------------------------------------------------------------------
__global__ void __launch_bounds__(128)
esn_xin(const float* __restrict__ x, const float* __restrict__ wscale,
        const unsigned* __restrict__ wih, const unsigned* __restrict__ wil,
        float* __restrict__ out) {
  const int lane  = threadIdx.x & 31;
  const int wave  = threadIdx.x >> 5;
  const int mtile = blockIdx.x;                       // 0..1021
  const int r = mtile * 16 + (lane & 15);             // global row 0..16351
  const int b = r / 511;
  const int t = 1 + (r - b * 511);
  const float* arow = x + (size_t)(b * 512 + t) * 128;
  const int khsel = (lane >> 4) << 3;

  // Build A fragments once (in registers).
  v8i ahv[4], alv[4];
#pragma unroll
  for (int kb = 0; kb < 4; ++kb) {
#pragma unroll
    for (int v = 0; v < 8; ++v) {
      const int k = kb * 32 + (v & 3) * 2 + ((v >> 2) << 4) + khsel;
      v2f f = *(const v2f*)(arow + k);
      const float h0 = (float)(__bf16)f.x;
      const float h1 = (float)(__bf16)f.y;
      ahv[kb][v] = (int)pack_bf16_pair(f.x, f.y);
      alv[kb][v] = (int)pack_bf16_pair(f.x - h0, f.y - h1);
    }
  }

  // Output row offsets (without column) for the 8 C rows this lane holds.
  size_t rowoff[8];
#pragma unroll
  for (int i = 0; i < 8; ++i) {
    const int rr = mtile * 16 + i + ((lane >> 4) << 3);
    const int bb = rr / 511;
    const int tt = 1 + (rr - bb * 511);
    rowoff[i] = (size_t)(bb * 512 + tt) * 1024;
  }
  const float s = wscale[0];

  for (int nt = 0; nt < 16; ++nt) {
    const int ntile = wave * 16 + nt;                 // 0..63
    const size_t base = (size_t)ntile * 1024;         // 4 kblks * 256 dwords
    v8f acc = {};
#pragma unroll
    for (int kb = 0; kb < 4; ++kb) {
      const size_t fo = base + ((size_t)kb * 32 + lane) * 8;
      v16bf BH = __builtin_bit_cast(v16bf, *(const v8i*)(wih + fo));
      v16bf BL = __builtin_bit_cast(v16bf, *(const v8i*)(wil + fo));
      v16bf AH = __builtin_bit_cast(v16bf, ahv[kb]);
      v16bf AL = __builtin_bit_cast(v16bf, alv[kb]);
      acc = __builtin_amdgcn_wmma_f32_16x16x32_bf16(false, AH, false, BH,
                                                    (short)0, acc, false, false);
      acc = __builtin_amdgcn_wmma_f32_16x16x32_bf16(false, AH, false, BL,
                                                    (short)0, acc, false, false);
      acc = __builtin_amdgcn_wmma_f32_16x16x32_bf16(false, AL, false, BH,
                                                    (short)0, acc, false, false);
    }
    const int n = ntile * 16 + (lane & 15);
#pragma unroll
    for (int i = 0; i < 8; ++i) out[rowoff[i] + n] = acc[i] * s;
  }
}

// ---------------------------------------------------------------------------
// Recurrent scan: persistent kernel, 128 blocks x 128 threads.
// Block = one 16x16 output tile; 4 waves split K (8 kblks each, 24 WMMAs),
// LDS reduce, tanh+noise epilogue that ALSO emits next step's packed bf16
// hi/lo A fragments (double buffered), then grid-wide barrier.
// Inner loop is pure b128 loads + WMMA.
// ---------------------------------------------------------------------------
__global__ void __launch_bounds__(128, 1)
esn_recurrent(const float* __restrict__ noise,
              const unsigned* __restrict__ wrh, const unsigned* __restrict__ wrl,
              unsigned* __restrict__ ah0, unsigned* __restrict__ al0,
              unsigned* __restrict__ ah1, unsigned* __restrict__ al1,
              float* __restrict__ out, int* bar) {
  __shared__ float red[4 * 256];
  const int lane  = threadIdx.x & 31;
  const int wave  = threadIdx.x >> 5;
  const int mtile = blockIdx.x & 1;
  const int ntile = blockIdx.x >> 1;
  const unsigned* bh = wrh + (size_t)ntile * 8192;    // 32 kblks * 256 dwords
  const unsigned* bl = wrl + (size_t)ntile * 8192;
  int* cnt = bar;
  int* gen = bar + 1;
  const int nblocks = (int)gridDim.x;

  // Warm this wave's B fragments (L2-resident) into the near caches.
  for (int kb = wave * 8; kb < wave * 8 + 8; ++kb) {
    __builtin_prefetch(bh + ((size_t)kb * 32 + lane) * 8, 0, 3);
    __builtin_prefetch(bl + ((size_t)kb * 32 + lane) * 8, 0, 3);
  }

  for (int t = 1; t < 512; ++t) {
    const unsigned* aph = ((t & 1) ? ah0 : ah1) + (size_t)mtile * 8192;
    const unsigned* apl = ((t & 1) ? al0 : al1) + (size_t)mtile * 8192;

    v8f acc = {};
    for (int kb = wave * 8; kb < wave * 8 + 8; ++kb) {
      const size_t fo = ((size_t)kb * 32 + lane) * 8;
      v16bf AH = __builtin_bit_cast(v16bf, *(const v8i*)(aph + fo));
      v16bf AL = __builtin_bit_cast(v16bf, *(const v8i*)(apl + fo));
      v16bf BH = __builtin_bit_cast(v16bf, *(const v8i*)(bh + fo));
      v16bf BL = __builtin_bit_cast(v16bf, *(const v8i*)(bl + fo));
      acc = __builtin_amdgcn_wmma_f32_16x16x32_bf16(false, AH, false, BH,
                                                    (short)0, acc, false, false);
      acc = __builtin_amdgcn_wmma_f32_16x16x32_bf16(false, AH, false, BL,
                                                    (short)0, acc, false, false);
      acc = __builtin_amdgcn_wmma_f32_16x16x32_bf16(false, AL, false, BH,
                                                    (short)0, acc, false, false);
    }
#pragma unroll
    for (int i = 0; i < 8; ++i) {
      const int m = i + ((lane >> 4) << 3);
      red[wave * 256 + m * 16 + (lane & 15)] = acc[i];
    }
    __syncthreads();

    // Epilogue: pre + tanh + noise; write out AND next-step packed A frags.
    unsigned short* dsh = (unsigned short*)((t & 1) ? ah1 : ah0);
    unsigned short* dsl = (unsigned short*)((t & 1) ? al1 : al0);
    for (int e = threadIdx.x; e < 256; e += 128) {
      const int m = e >> 4, n = e & 15;
      const int b  = mtile * 16 + m;
      const int ng = ntile * 16 + n;
      const size_t o = (size_t)(b * 512 + t) * 1024 + ng;
      const float pre = red[e] + red[256 + e] + red[512 + e] + red[768 + e]
                        + out[o];  // xin staged here by esn_xin
      const float post = tanhf(pre) + NOISE_F * noise[(size_t)b * 1024 + ng];
      out[o] = post;
      // Packed A-frag position for element (row m of mtile, k = ng).
      const __bf16 hb = (__bf16)post;
      const __bf16 lb = (__bf16)(post - (float)hb);
      const int kb   = ng >> 5;
      const int koff = ng & 31;
      const int sub  = koff & 1;
      const int rem  = koff >> 1;
      const int v    = (rem & 3) | ((rem >> 3) << 2);
      const int half = (rem >> 2) & 1;
      const int la   = m | (half << 4);
      const size_t di = (((size_t)(mtile * 32 + kb) * 32 + la) << 3) + v;
      dsh[di * 2 + sub] = __builtin_bit_cast(unsigned short, hb);
      dsl[di * 2 + sub] = __builtin_bit_cast(unsigned short, lb);
    }
    __syncthreads();            // red[] reuse next iteration
    __threadfence();            // publish state + packed frags device-wide
    if (threadIdx.x == 0) {
      const int g = atomicAdd(gen, 0);
      __threadfence();
      if (atomicAdd(cnt, 1) == nblocks - 1) {
        atomicExch(cnt, 0);
        __threadfence();
        atomicExch(gen, g + 1);
      } else {
        while (atomicAdd(gen, 0) == g) __builtin_amdgcn_s_sleep(1);
      }
    }
    __syncthreads();
    __threadfence();            // acquire other blocks' writes
  }
}

// ---------------------------------------------------------------------------
extern "C" void kernel_launch(void* const* d_in, const int* in_sizes, int n_in,
                              void* d_out, int out_size, void* d_ws, size_t ws_size,
                              hipStream_t stream) {
  const float* x       = (const float*)d_in[0];  // [32,512,128]
  const float* w_input = (const float*)d_in[1];  // [128,1024]
  const float* w_res   = (const float*)d_in[2];  // [1024,1024]
  const float* wscale  = (const float*)d_in[3];  // scalar
  const float* states0 = (const float*)d_in[4];  // [32,1024]
  const float* step0   = (const float*)d_in[5];  // [32,1024]
  const float* noise   = (const float*)d_in[6];  // [32,1024]
  float* out = (float*)d_out;                    // [32,512,1024]

  // Workspace layout (needs ~4.75 MB + 8 B):
  unsigned char* ws = (unsigned char*)d_ws;
  unsigned* wrh = (unsigned*)(ws);                                  // 2 MB
  unsigned* wrl = (unsigned*)(ws + (2u << 20));                     // 2 MB
  unsigned* wih = (unsigned*)(ws + (4u << 20));                     // 256 KB
  unsigned* wil = (unsigned*)(ws + (4u << 20) + (256u << 10));      // 256 KB
  unsigned* ah0 = (unsigned*)(ws + (4u << 20) + (512u << 10));      // 64 KB
  unsigned* al0 = (unsigned*)(ws + (4u << 20) + (576u << 10));      // 64 KB
  unsigned* ah1 = (unsigned*)(ws + (4u << 20) + (640u << 10));      // 64 KB
  unsigned* al1 = (unsigned*)(ws + (4u << 20) + (704u << 10));      // 64 KB
  int* bar      = (int*)(ws + (4u << 20) + (768u << 10));           // 8 B

  // 1) split+swizzle weights (also resets the global barrier)
  esn_pack<<<2048, 256, 0, stream>>>(w_res, 1024, 1024, wrh, wrl, bar);
  esn_pack<<<256, 256, 0, stream>>>(w_input, 128, 1024, wih, wil, nullptr);
  // 2) pack states0 into phase-0 A fragments
  esn_pack_s0<<<64, 256, 0, stream>>>(states0, ah0, al0);
  // 3) out[:,0,:] = step0
  esn_t0<<<128, 256, 0, stream>>>(step0, out);
  // 4) out[:,1:,:] = scale * x[:,1:,:] @ w_input   (WMMA, A in registers)
  esn_xin<<<1022, 128, 0, stream>>>(x, wscale, wih, wil, out);
  // 5) persistent recurrent scan (pure load+WMMA inner loop)
  esn_recurrent<<<128, 128, 0, stream>>>(noise, wrh, wrl, ah0, al0, ah1, al1,
                                         out, bar);
}
